// HistoGAN_71871982731636
// MI455X (gfx1250) — matrix operand order repaired and checked by reading, not compiled, not run on hardware
//
#include <hip/hip_runtime.h>
#include <hip/hip_bf16.h>

typedef __attribute__((ext_vector_type(16))) __bf16 v16bf;
typedef __attribute__((ext_vector_type(8)))  __bf16 v8bf;
typedef __attribute__((ext_vector_type(2)))  __bf16 v2bf;
typedef __attribute__((ext_vector_type(8)))  float  v8f;

__device__ __forceinline__ __bf16 f2bf(float f) {
  // round-to-nearest-even fp32 -> bf16
  unsigned u = __float_as_uint(f);
  u += 0x7fffu + ((u >> 16) & 1u);
  unsigned short s = (unsigned short)(u >> 16);
  return __builtin_bit_cast(__bf16, s);
}

__device__ __forceinline__ int imin(int a, int b) { return a < b ? a : b; }
__device__ __forceinline__ int imax(int a, int b) { return a > b ? a : b; }

__device__ __forceinline__ v16bf cat8(v8bf lo, v8bf hi) {
  return __builtin_shufflevector(lo, hi, 0, 1, 2, 3, 4, 5, 6, 7,
                                 8, 9, 10, 11, 12, 13, 14, 15);
}

// ---------------------------------------------------------------------------
// GEMM: Y[M,N] = act( X[M,K] @ W[N,K]^T + bias[N] ),  X/W bf16, acc f32.
// Each wave computes NPW 16x16 tiles sharing one A fragment:
//   per K-step: 2 x b128 (A) + NPW*2 x b128 (B) + NPW x WMMA.
// ---------------------------------------------------------------------------
template <int NPW>
__global__ void gemm_bf16_wmma(const __bf16* __restrict__ X,
                               const __bf16* __restrict__ W,
                               const float* __restrict__ bias,
                               float* __restrict__ Yf,
                               __bf16* __restrict__ Ybf,
                               int M, int N, int K, int act,
                               int ngroups, int total) {
  int wave = threadIdx.x >> 5;
  int lane = threadIdx.x & 31;
  int tid  = blockIdx.x * (blockDim.x >> 5) + wave;
  if (tid >= total) return;                 // wave-uniform: EXEC all-1 for WMMA
  int mt = tid / ngroups, ng = tid % ngroups;
  int half = lane >> 4, l16 = lane & 15;
  int m0 = mt * 16, nb0 = ng * NPW * 16;

  const __bf16* xr = X + (size_t)(m0 + l16) * K + 8 * half;
  const __bf16* wr = W + (size_t)(nb0 + l16) * K + 8 * half;

  v8f acc[NPW];
#pragma unroll
  for (int j = 0; j < NPW; ++j) acc[j] = (v8f){};

  for (int k0 = 0; k0 < K; k0 += 32) {
    __builtin_prefetch(wr + k0 + 128, 0, 1);
    v8bf a0 = *(const v8bf*)(xr + k0);
    v8bf a1 = *(const v8bf*)(xr + k0 + 16);
    v16bf av = cat8(a0, a1);
#pragma unroll
    for (int j = 0; j < NPW; ++j) {
      const __bf16* wj = wr + (size_t)j * 16 * K + k0;
      v8bf b0 = *(const v8bf*)(wj);
      v8bf b1 = *(const v8bf*)(wj + 16);
      acc[j] = __builtin_amdgcn_wmma_f32_16x16x32_bf16(false, av, false, cat8(b0, b1),
                                                       (short)0, acc[j], false, false);
    }
  }
#pragma unroll
  for (int j = 0; j < NPW; ++j) {
    int n = nb0 + j * 16 + l16;
    float bv = bias ? bias[n] : 0.0f;
#pragma unroll
    for (int v = 0; v < 8; ++v) {
      int m = m0 + v + 8 * half;
      float y = acc[j][v] + bv;
      if (act) y = (y >= 0.0f) ? y : 0.2f * y;
      if (Yf)  Yf[(size_t)m * N + n] = y;
      if (Ybf) Ybf[(size_t)m * N + n] = f2bf(y);
    }
  }
}

// ---------------------------------------------------------------------------
// Mod/demod 3x3 conv as implicit GEMM over pre-modulated, pair-interleaved,
// halo-padded bf16 input:
//   xm[b][c/2][y+1][x+1][2] = bf16(s1[b,c]*x[b,c,y,x])  (halo = 0)
// K = 9*C, k = tap*C + c.  A element pair (c,c+1)@pixel = one aligned b32 load,
// no bounds checks (padding absorbs dy-1/dx-1).  Wp bf16 [O,9C] (B contiguous).
// NPW n-tiles per wave share one A fragment.
// ---------------------------------------------------------------------------
template <int NPW>
__global__ void conv_mod3x3_wmma(const v2bf* __restrict__ xm,      // [B,C/2,H+2,W+2]
                                 const __bf16* __restrict__ Wp,    // [O,9C]
                                 const float* __restrict__ isg,    // [B,O]
                                 const float* __restrict__ noise,  // [B,H*W]
                                 const float* __restrict__ bias,   // [O]
                                 const float* __restrict__ nsf,    // [1]
                                 float* __restrict__ out,          // [B,O,H,W]
                                 int C, int logC, int H, int W, int logW, int O,
                                 int mtiles, int ngroups, int total) {
  int wave = threadIdx.x >> 5;
  int lane = threadIdx.x & 31;
  int tid  = blockIdx.x * (blockDim.x >> 5) + wave;
  if (tid >= total) return;
  int b  = tid / (mtiles * ngroups);
  int r  = tid % (mtiles * ngroups);
  int mt = r / ngroups, ng = r % ngroups;
  int half = lane >> 4, l16 = lane & 15;
  int HW = H * W, PW = W + 2, PH = H + 2;
  int p  = mt * 16 + l16;                  // A-row output pixel
  int oy = p >> logW, ox = p & (W - 1);
  int nb0 = ng * NPW * 16;
  int K  = 9 * C;

  const v2bf*   xb = xm + (size_t)b * (C >> 1) * PH * PW;
  const __bf16* wr = Wp + (size_t)(nb0 + l16) * K + 8 * half;

  v8f acc[NPW];
#pragma unroll
  for (int j = 0; j < NPW; ++j) acc[j] = (v8f){};

  for (int k0 = 0; k0 < K; k0 += 32) {
    __builtin_prefetch(wr + k0 + 128, 0, 1);
    v16bf a;
#pragma unroll
    for (int v = 0; v < 8; ++v) {
      int kb  = k0 + ((v < 4) ? 2 * v : 16 + 2 * (v - 4)) + 8 * half;
      int c2  = (kb & (C - 1)) >> 1;       // channel pair
      int tap = kb >> logC;
      int ty  = tap / 3, tx = tap % 3;     // padded offsets (dy-1+1, dx-1+1)
      v2bf pv = xb[((size_t)c2 * PH + (oy + ty)) * PW + (ox + tx)];
      a[2 * v]     = pv[0];
      a[2 * v + 1] = pv[1];
    }
#pragma unroll
    for (int j = 0; j < NPW; ++j) {
      const __bf16* wj = wr + (size_t)j * 16 * K + k0;
      v8bf b0 = *(const v8bf*)(wj);
      v8bf b1 = *(const v8bf*)(wj + 16);
      acc[j] = __builtin_amdgcn_wmma_f32_16x16x32_bf16(false, a, false, cat8(b0, b1),
                                                       (short)0, acc[j], false, false);
    }
  }
  float ns = nsf[0];
  const float* nb = noise + (size_t)b * HW;
#pragma unroll
  for (int j = 0; j < NPW; ++j) {
    int n = nb0 + j * 16 + l16;
    float is = isg[(size_t)b * O + n];
    float bv = bias[n];
    float* ob = out + ((size_t)b * O + n) * HW;
#pragma unroll
    for (int v = 0; v < 8; ++v) {
      int m = mt * 16 + v + 8 * half;
      float y = acc[j][v] * is + ns * nb[m] + bv;
      ob[m] = (y >= 0.0f) ? y : 0.2f * y;
    }
  }
}

// ---- streaming helper kernels --------------------------------------------

__global__ void f32_to_bf16_k(const float* __restrict__ in, __bf16* __restrict__ out, int n) {
  int i = blockIdx.x * blockDim.x + threadIdx.x;
  if (i < n) out[i] = f2bf(in[i]);
}

// Pair-interleaved, halo-padded premod: xm[b][c2][y][x] = {s*x(c=2c2), s*x(c=2c2+1)}
__global__ void premod_pad_bf16(const float* __restrict__ x, const float* __restrict__ s1,
                                v2bf* __restrict__ xm, int C, int H, int W, int total) {
  int i = blockIdx.x * blockDim.x + threadIdx.x;
  if (i >= total) return;
  int PW = W + 2, PH = H + 2;
  int pp = i % (PH * PW);
  int c2 = (i / (PH * PW)) % (C >> 1);
  int b  = i / ((PH * PW) * (C >> 1));
  int y = pp / PW, xq = pp % PW;
  v2bf o = {};
  if (y >= 1 && y <= H && xq >= 1 && xq <= W) {
    size_t base = ((size_t)b * C + 2 * c2) * H * W + (size_t)(y - 1) * W + (xq - 1);
    float s0 = s1[(size_t)b * C + 2 * c2];
    float s1v = s1[(size_t)b * C + 2 * c2 + 1];
    o[0] = f2bf(x[base] * s0);
    o[1] = f2bf(x[base + (size_t)H * W] * s1v);
  }
  xm[i] = o;
}

// wsq[o*C+c] = sum over 3x3 taps of w^2
__global__ void wsq_kernel(const float* __restrict__ w, float* __restrict__ wsq, int OC) {
  int i = blockIdx.x * blockDim.x + threadIdx.x;
  if (i >= OC) return;
  const float* p = w + (size_t)i * 9;
  float s = 0.0f;
#pragma unroll
  for (int t = 0; t < 9; ++t) { float v = p[t]; s += v * v; }
  wsq[i] = s;
}

// isg[b,o] = rsqrt( sum_c s1[b,c]^2 * wsq[o,c] + 1e-8 )
__global__ void invsig_kernel(const float* __restrict__ s1, const float* __restrict__ wsq,
                              float* __restrict__ isg, int B, int C, int O) {
  int i = blockIdx.x * blockDim.x + threadIdx.x;
  if (i >= B * O) return;
  int b = i / O, o = i % O;
  const float* sb = s1 + (size_t)b * C;
  const float* wb = wsq + (size_t)o * C;
  float acc = 0.0f;
  for (int c = 0; c < C; ++c) { float sv = sb[c]; acc += sv * sv * wb[c]; }
  isg[i] = rsqrtf(acc + 1e-8f);
}

// Wp[o*9C + tap*C + c] = bf16( w[o,c,tap] )
__global__ void pack_w3x3(const float* __restrict__ w, __bf16* __restrict__ Wp, int O, int C) {
  int i = blockIdx.x * blockDim.x + threadIdx.x;
  int total = O * C * 9;
  if (i >= total) return;
  int o = i / (C * 9), r = i % (C * 9), c = r / 9, t = r % 9;
  Wp[(size_t)o * 9 * C + (size_t)t * C + c] = f2bf(w[i]);
}

// tRGB (mod, no demod): out[b,o,p] = sum_c s3[b,c]*wrgb[o,c]*h[b,c,p], o in 0..2
__global__ void rgb_mod1x1(const float* __restrict__ h, const float* __restrict__ s3,
                           const float* __restrict__ wrgb, float* __restrict__ out,
                           int C, int HW, int total) {
  int i = blockIdx.x * blockDim.x + threadIdx.x;
  if (i >= total) return;
  int b = i / HW, p = i % HW;
  const float* hb = h + (size_t)b * C * HW + p;
  const float* sb = s3 + (size_t)b * C;
  float a0 = 0.f, a1 = 0.f, a2 = 0.f;
  for (int c = 0; c < C; ++c) {
    float m = hb[(size_t)c * HW] * sb[c];
    a0 += m * wrgb[c];
    a1 += m * wrgb[C + c];
    a2 += m * wrgb[2 * C + c];
  }
  float* ob = out + (size_t)b * 3 * HW + p;
  ob[0] = a0; ob[HW] = a1; ob[2 * HW] = a2;
}

// Bilinear 2x upsample (half-pixel centers, edge clamp == jax 'bilinear').
// If b2 != nullptr, upsamples (a + b2) fused.
__global__ void upsample2x_sum(const float* __restrict__ a, const float* __restrict__ b2,
                               float* __restrict__ out, int H, int W, int total) {
  int i = blockIdx.x * blockDim.x + threadIdx.x;
  if (i >= total) return;
  int OW = 2 * W, OH = 2 * H;
  int bc = i / (OH * OW), r = i % (OH * OW);
  int oy = r / OW, ox = r % OW;
  float fy = (oy + 0.5f) * 0.5f - 0.5f;
  float fx = (ox + 0.5f) * 0.5f - 0.5f;
  int y0 = (int)floorf(fy); float wy = fy - (float)y0;
  int x0 = (int)floorf(fx); float wx = fx - (float)x0;
  int y0c = imin(imax(y0, 0), H - 1), y1c = imin(imax(y0 + 1, 0), H - 1);
  int x0c = imin(imax(x0, 0), W - 1), x1c = imin(imax(x0 + 1, 0), W - 1);
  const float* pa = a + (size_t)bc * H * W;
  const float* pb = b2 ? b2 + (size_t)bc * H * W : nullptr;
  auto ld = [&](int y, int x) -> float {
    float v = pa[y * W + x];
    if (pb) v += pb[y * W + x];
    return v;
  };
  out[i] = (1.f - wy) * ((1.f - wx) * ld(y0c, x0c) + wx * ld(y0c, x1c)) +
           wy         * ((1.f - wx) * ld(y1c, x0c) + wx * ld(y1c, x1c));
}

__global__ void bcast_const(const float* __restrict__ c, float* __restrict__ out,
                            int n, int total) {
  int i = blockIdx.x * blockDim.x + threadIdx.x;
  if (i >= total) return;
  out[i] = c[i % n];
}

__global__ void add2_kernel(const float* __restrict__ a, const float* __restrict__ b,
                            float* __restrict__ o, int n) {
  int i = blockIdx.x * blockDim.x + threadIdx.x;
  if (i < n) o[i] = a[i] + b[i];
}

__global__ void copy_kernel(const float* __restrict__ a, float* __restrict__ o, int n) {
  int i = blockIdx.x * blockDim.x + threadIdx.x;
  if (i < n) o[i] = a[i];
}

// ---------------------------------------------------------------------------
// Host orchestration
// ---------------------------------------------------------------------------
static inline int ilog2(int v) { int l = 0; while ((1 << l) < v) ++l; return l; }

extern "C" void kernel_launch(void* const* d_in, const int* in_sizes, int n_in,
                              void* d_out, int out_size, void* d_ws, size_t ws_size,
                              hipStream_t stream) {
  (void)in_sizes; (void)n_in; (void)out_size; (void)ws_size;
  const int B = 128;
  const int IN_CH[5]  = {64, 512, 256, 128, 64};
  const int OUT_CH[5] = {512, 256, 128, 64, 32};
  const int RES[5]    = {4, 8, 16, 32, 64};

  // ---- unpack inputs (dict insertion order) ----
  int ix = 0;
  const float* z     = (const float*)d_in[ix++];
  const float* thist = (const float*)d_in[ix++];
  const float *lm_w[8], *lm_b[8];
  for (int i = 0; i < 8; ++i) { lm_w[i] = (const float*)d_in[ix++]; lm_b[i] = (const float*)d_in[ix++]; }
  const float *hp_w[8], *hp_b[8];
  for (int i = 0; i < 8; ++i) { hp_w[i] = (const float*)d_in[ix++]; hp_b[i] = (const float*)d_in[ix++]; }
  const float* cconst = (const float*)d_in[ix++];
  struct Blk { const float *a1w,*a1b,*a2w,*a2b,*a3w,*a3b,*w1,*w2,*wrgb,*nsf1,*nsf2,*b1,*b2; };
  Blk blk[5];
  for (int i = 0; i < 5; ++i) {
    blk[i].a1w  = (const float*)d_in[ix++]; blk[i].a1b  = (const float*)d_in[ix++];
    blk[i].a2w  = (const float*)d_in[ix++]; blk[i].a2b  = (const float*)d_in[ix++];
    blk[i].a3w  = (const float*)d_in[ix++]; blk[i].a3b  = (const float*)d_in[ix++];
    blk[i].w1   = (const float*)d_in[ix++]; blk[i].w2   = (const float*)d_in[ix++];
    blk[i].wrgb = (const float*)d_in[ix++];
    blk[i].nsf1 = (const float*)d_in[ix++]; blk[i].nsf2 = (const float*)d_in[ix++];
    blk[i].b1   = (const float*)d_in[ix++]; blk[i].b2   = (const float*)d_in[ix++];
  }
  const float* noise[5][2];
  for (int i = 0; i < 5; ++i) { noise[i][0] = (const float*)d_in[ix++]; noise[i][1] = (const float*)d_in[ix++]; }

  // ---- workspace layout ----
  char* wsp = (char*)d_ws;
  size_t off = 0;
  auto alloc = [&](size_t bytes) -> char* {
    char* p = wsp + off; off += (bytes + 255) & ~(size_t)255; return p;
  };
  const size_t FM_MAX  = (size_t)B * 64 * 64 * 64;           // peak fm elems (33.5M)
  const size_t XM_MAX  = (size_t)B * 32 * 66 * 66;           // peak padded pair-dwords
  float*  fmA   = (float*)alloc(FM_MAX * 4);
  float*  fmB   = (float*)alloc(FM_MAX * 4);
  v2bf*   xmod  = (v2bf*)alloc(XM_MAX * 4);                  // premodulated padded input
  float*  rgbS0 = (float*)alloc((size_t)B * 3 * 64 * 64 * 4);
  float*  rgbS1 = (float*)alloc((size_t)B * 3 * 64 * 64 * 4);
  float*  rgbC  = (float*)alloc((size_t)B * 3 * 64 * 64 * 4);
  __bf16* zbf   = (__bf16*)alloc((size_t)B * 512 * 2);
  __bf16* thbf  = (__bf16*)alloc((size_t)B * 12288 * 2);
  __bf16* abf0  = (__bf16*)alloc((size_t)B * 1024 * 2);
  __bf16* abf1  = (__bf16*)alloc((size_t)B * 1024 * 2);
  __bf16* wlatb = (__bf16*)alloc((size_t)B * 512 * 2);
  __bf16* histb = (__bf16*)alloc((size_t)B * 512 * 2);
  float*  wlatf = (float*)alloc((size_t)B * 512 * 4);
  float*  s1b   = (float*)alloc((size_t)B * 512 * 4);
  float*  s2b   = (float*)alloc((size_t)B * 512 * 4);
  float*  s3b   = (float*)alloc((size_t)B * 512 * 4);
  float*  isg   = (float*)alloc((size_t)B * 512 * 4);
  float*  wsq   = (float*)alloc((size_t)512 * 512 * 4);
  __bf16* wbuf  = (__bf16*)alloc((size_t)1024 * 12288 * 2);
  __bf16* Wpack = (__bf16*)alloc((size_t)512 * 9 * 512 * 2);

  auto cdiv = [](int a, int b) { return (a + b - 1) / b; };

  auto wcvt = [&](const float* w, int n) {
    f32_to_bf16_k<<<cdiv(n, 256), 256, 0, stream>>>(w, wbuf, n);
  };
  auto gemm = [&](const __bf16* X, const float* bias, float* Yf, __bf16* Ybf,
                  int M, int N, int K, int act) {
    int ng = N / 32, total = (M / 16) * ng;   // NPW = 2
    gemm_bf16_wmma<2><<<cdiv(total, 8), 256, 0, stream>>>(X, wbuf, bias, Yf, Ybf,
                                                          M, N, K, act, ng, total);
  };

  auto conv = [&](const float* x, const float* w_raw, const float* s,
                  const float* nsfp, const float* noise_p, const float* bias_p,
                  float* out, int C, int R, int O) {
    int OC = O * C, HW = R * R;
    pack_w3x3<<<cdiv(OC * 9, 256), 256, 0, stream>>>(w_raw, Wpack, O, C);
    wsq_kernel<<<cdiv(OC, 256), 256, 0, stream>>>(w_raw, wsq, OC);
    invsig_kernel<<<cdiv(B * O, 256), 256, 0, stream>>>(s, wsq, isg, B, C, O);
    int npad = B * (C >> 1) * (R + 2) * (R + 2);
    premod_pad_bf16<<<cdiv(npad, 256), 256, 0, stream>>>(x, s, xmod, C, R, R, npad);
    int mt = HW / 16, ntiles = O / 16;
    if (ntiles % 4 == 0) {
      int ng = ntiles / 4, total = B * mt * ng;
      conv_mod3x3_wmma<4><<<cdiv(total, 8), 256, 0, stream>>>(
          xmod, Wpack, isg, noise_p, bias_p, nsfp, out,
          C, ilog2(C), R, R, ilog2(R), O, mt, ng, total);
    } else {
      int ng = ntiles / 2, total = B * mt * ng;
      conv_mod3x3_wmma<2><<<cdiv(total, 8), 256, 0, stream>>>(
          xmod, Wpack, isg, noise_p, bias_p, nsfp, out,
          C, ilog2(C), R, R, ilog2(R), O, mt, ng, total);
    }
  };

  // ---- input conversions ----
  f32_to_bf16_k<<<cdiv(B * 512, 256), 256, 0, stream>>>(z, zbf, B * 512);
  f32_to_bf16_k<<<cdiv(B * 12288, 256), 256, 0, stream>>>(thist, thbf, B * 12288);

  // ---- latent mapping MLP: 8x (512->512, lrelu) ----
  {
    const __bf16* X = zbf;
    for (int i = 0; i < 8; ++i) {
      wcvt(lm_w[i], 512 * 512);
      __bf16* Yb = (i == 7) ? wlatb : ((i & 1) ? abf1 : abf0);
      float*  Yf = (i == 7) ? wlatf : nullptr;
      gemm(X, lm_b[i], Yf, Yb, B, 512, 512, 1);
      X = Yb;
    }
  }

  // ---- histogram MLP: 12288->1024->512->512x6, lrelu ----
  wcvt(hp_w[0], 1024 * 12288);
  gemm(thbf, hp_b[0], nullptr, abf0, B, 1024, 12288, 1);
  wcvt(hp_w[1], 512 * 1024);
  gemm(abf0, hp_b[1], nullptr, abf1, B, 512, 1024, 1);
  {
    const __bf16* X = abf1;
    for (int i = 2; i < 8; ++i) {
      wcvt(hp_w[i], 512 * 512);
      __bf16* Yb = (i == 7) ? histb : ((i & 1) ? abf1 : abf0);
      gemm(X, hp_b[i], nullptr, Yb, B, 512, 512, 1);
      X = Yb;
    }
  }

  // ---- synthesis blocks ----
  bcast_const<<<cdiv(B * 64 * 16, 256), 256, 0, stream>>>(cconst, fmA, 64 * 16, B * 64 * 16);
  float* cur = fmA;
  float* alt = fmB;
  float* rsum = rgbS0;
  float* rtmp = rgbS1;

  for (int i = 0; i < 5; ++i) {
    int C = IN_CH[i], O = OUT_CH[i], R = RES[i], HW = R * R;
    const __bf16* wv = (i < 4) ? wlatb : histb;
    wcvt(blk[i].a1w, C * 512);  gemm(wv, blk[i].a1b, s1b, nullptr, B, C, 512, 0);
    conv(cur, blk[i].w1, s1b, blk[i].nsf1, noise[i][0], blk[i].b1, alt, C, R, O); // h1 -> alt
    wcvt(blk[i].a2w, O * 512);  gemm(wv, blk[i].a2b, s2b, nullptr, B, O, 512, 0);
    conv(alt, blk[i].w2, s2b, blk[i].nsf2, noise[i][1], blk[i].b2, cur, O, R, O); // h2 -> cur
    wcvt(blk[i].a3w, O * 512);  gemm(wv, blk[i].a3b, s3b, nullptr, B, O, 512, 0);
    rgb_mod1x1<<<cdiv(B * HW, 256), 256, 0, stream>>>(cur, s3b, blk[i].wrgb, rgbC, O, HW, B * HW);
    if (i < 4) {
      upsample2x_sum<<<cdiv(B * O * 4 * HW, 256), 256, 0, stream>>>(
          cur, nullptr, alt, R, R, B * O * 4 * HW);
      upsample2x_sum<<<cdiv(B * 3 * 4 * HW, 256), 256, 0, stream>>>(
          (i == 0) ? rgbC : rsum, (i == 0) ? nullptr : rgbC, rtmp, R, R, B * 3 * 4 * HW);
      { float* t = rsum; rsum = rtmp; rtmp = t; }
      { float* t = cur; cur = alt; alt = t; }
    }
  }

  // ---- outputs: [rgb_sum + rgb (128*3*64*64)] ++ [wlat (128*512)] ----
  float* outp = (float*)d_out;
  int nrgb = B * 3 * 64 * 64;
  add2_kernel<<<cdiv(nrgb, 256), 256, 0, stream>>>(rsum, rgbC, outp, nrgb);
  copy_kernel<<<cdiv(B * 512, 256), 256, 0, stream>>>(wlatf, outp + nrgb, B * 512);
}
